// DKRL_60344290509522
// MI455X (gfx1250) — matrix operand order repaired and checked by reading, not compiled, not run on hardware
//
#include <hip/hip_runtime.h>
#include <hip/hip_bf16.h>

// DKRL forward, fused for gfx1250 (MI455X), wave32 + WMMA f16->f32.

typedef _Float16 v16h __attribute__((ext_vector_type(16)));
typedef _Float16 v4h  __attribute__((ext_vector_type(4)));
typedef float    v8f  __attribute__((ext_vector_type(8)));

#define L_SEQ   343
#define NW      100
#define N1      100
#define N2      100
#define KPAD    128     // padded K for both GEMMs (4 chunks of 32)
#define NPAD    112     // padded N (7 tiles of 16)
#define P_REAL  86      // pooled length
#define HROWS   96      // pooled rows padded to 6 M-tiles of 16
#define LTILES  22      // ceil(343/16)
#define FTILES  7       // 112/16
#define MTILES2 6       // 96/16

// ---------------------------------------------------------------------------
// Prep: W1_eff[n][k] = W_1[n][k] + W_1[n][100+k]; convert W1_eff, W_2 to f16,
// zero-padded to [112][128].
// ---------------------------------------------------------------------------
__global__ void dkrl_prep(const float* __restrict__ W1,
                          const float* __restrict__ W2,
                          _Float16* __restrict__ w1e,
                          _Float16* __restrict__ w2e) {
    int i = blockIdx.x * blockDim.x + threadIdx.x;
    if (i >= NPAD * KPAD) return;
    int n = i >> 7;
    int k = i & (KPAD - 1);
    float v1 = 0.f, v2 = 0.f;
    if (n < N1 && k < NW) {
        v1 = W1[n * (2 * NW) + k] + W1[n * (2 * NW) + NW + k];
        v2 = W2[n * N1 + k];
    }
    w1e[i] = (_Float16)v1;
    w2e[i] = (_Float16)v2;
}

// ---------------------------------------------------------------------------
// Fused main kernel: one block per batch row. 4 waves (128 threads).
//   Phase 1: gather emb -> f16 LDS, WMMA GEMM1, maxpool+relu -> h1 LDS (f16)
//   Phase 2: WMMA GEMM2 from h1 LDS, row-sum D-fragments -> mean -> out
// ---------------------------------------------------------------------------
__global__ __launch_bounds__(128) void dkrl_main(
    const int*   __restrict__ ent_words,   // [B, 343]
    const float* __restrict__ word_vec,    // [40000, 100]
    const _Float16* __restrict__ w1e,      // [112][128] f16
    const _Float16* __restrict__ w2e,      // [112][128] f16
    float* __restrict__ out)               // [B, 100]
{
    __shared__ __align__(16) _Float16 h1[HROWS * KPAD];    // 24 KB pooled acts
    __shared__ __align__(16) _Float16 aBuf[4][16 * KPAD];  // 16 KB A staging

    const int b    = blockIdx.x;
    const int tid  = threadIdx.x;
    const int wave = tid >> 5;
    const int lane = tid & 31;
    const int nrow = lane & 15;
    const int hiA  = (lane & 16) ? 8  : 0;   // A-fragment K interleave base
    const int hiB  = (lane & 16) ? 16 : 0;   // B-fragment K split base

    // zero h1 (pad rows 88..95, pad cols 112..127) and all of aBuf
    // (its k>=100 pad columns are never rewritten and must be exactly 0).
    {
        int4* z1 = (int4*)h1;
        for (int i = tid; i < (HROWS * KPAD) / 8; i += 128)
            z1[i] = make_int4(0, 0, 0, 0);
        int4* z2 = (int4*)(&aBuf[0][0]);
        for (int i = tid; i < (4 * 16 * KPAD) / 8; i += 128)
            z2[i] = make_int4(0, 0, 0, 0);
    }
    __syncthreads();

    // ---------------- Phase 1: GEMM1 + maxpool + relu ----------------
    for (int lt = wave; lt < LTILES; lt += 4) {
        // One coalesced load of the 16 word ids for this l-tile.
        // Pad rows are clamped to l=342: harmless for pool group p=85
        // (max(a,b,c,c)==max(a,b,c)); pooled rows p>=86 are zeroed at store.
        int lclamp = lt * 16 + nrow;
        lclamp = (lclamp < L_SEQ) ? lclamp : (L_SEQ - 1);
        const int widv = ent_words[b * L_SEQ + lclamp];

        // Gather 16 embedding rows, fp32 -> fp16, into per-wave A staging.
        // Lanes 0..24 each own one 4-float chunk column (16B global loads).
        const int ck = lane * 4;
        if (lane < 25) {
            #pragma unroll
            for (int r = 0; r < 16; ++r) {
                int w = __builtin_amdgcn_readlane(widv, r);  // SGPR row base
                float4 v = *(const float4*)(word_vec + (long)w * NW + ck);
                v4h hv;
                hv[0] = (_Float16)v.x;
                hv[1] = (_Float16)v.y;
                hv[2] = (_Float16)v.z;
                hv[3] = (_Float16)v.w;
                *(v4h*)(&aBuf[wave][r * KPAD + ck]) = hv;    // ds_store_b64
            }
        }
        asm volatile("s_wait_dscnt 0" ::: "memory");

        // load all 4 A K-chunks for this l-tile (kept in VGPRs across f-tiles)
        v16h a[4];
        #pragma unroll
        for (int c = 0; c < 4; ++c) {
            #pragma unroll
            for (int h = 0; h < 16; ++h) {
                int k = c * 32 + hiA + ((h < 8) ? h : h + 8);
                a[c][h] = aBuf[wave][nrow * KPAD + k];
            }
        }

        const int p0 = lt * 4 + ((lane & 16) ? 2 : 0);
        for (int ft = 0; ft < FTILES; ++ft) {
            v8f acc = {0.f, 0.f, 0.f, 0.f, 0.f, 0.f, 0.f, 0.f};
            #pragma unroll
            for (int c = 0; c < 4; ++c) {
                v16h bf;
                #pragma unroll
                for (int h = 0; h < 16; ++h)
                    bf[h] = w1e[(ft * 16 + nrow) * KPAD + c * 32 + hiA + ((h < 8) ? h : h + 8)];
                acc = __builtin_amdgcn_wmma_f32_16x16x32_f16(
                          false, a[c], false, bf, (short)0, acc, false, false);
            }
            // max-pool over 4 consecutive l rows + ReLU, directly on D layout:
            // lanes 0-15 hold M=vgpr, lanes 16-31 hold M=8+vgpr.
            float m0 = fmaxf(fmaxf(acc[0], acc[1]), fmaxf(acc[2], acc[3]));
            float m1 = fmaxf(fmaxf(acc[4], acc[5]), fmaxf(acc[6], acc[7]));
            m0 = fmaxf(m0, 0.f);
            m1 = fmaxf(m1, 0.f);
            // pooled rows beyond P_REAL are padding -> force exact zero
            m0 = (p0     < P_REAL) ? m0 : 0.f;
            m1 = (p0 + 1 < P_REAL) ? m1 : 0.f;
            int fcol = ft * 16 + nrow;
            h1[p0 * KPAD + fcol]       = (_Float16)m0;
            h1[(p0 + 1) * KPAD + fcol] = (_Float16)m1;
        }
    }
    __syncthreads();

    // ---------------- Phase 2: GEMM2 + mean over pooled positions ----------
    for (int nt = wave; nt < FTILES; nt += 4) {
        float nsum = 0.f;
        for (int mt = 0; mt < MTILES2; ++mt) {
            v8f acc = {0.f, 0.f, 0.f, 0.f, 0.f, 0.f, 0.f, 0.f};
            #pragma unroll
            for (int c = 0; c < 4; ++c) {
                v16h af, bf;
                #pragma unroll
                for (int h = 0; h < 16; ++h) {
                    int k = c * 32 + hiA + ((h < 8) ? h : h + 8);
                    af[h] = h1[(mt * 16 + nrow) * KPAD + k];
                }
                #pragma unroll
                for (int h = 0; h < 16; ++h)
                    bf[h] = w2e[(nt * 16 + nrow) * KPAD + c * 32 + hiA + ((h < 8) ? h : h + 8)];
                acc = __builtin_amdgcn_wmma_f32_16x16x32_f16(
                          false, af, false, bf, (short)0, acc, false, false);
            }
            // sum over the 16 M rows of this D tile (mean numerator)
            nsum += acc[0] + acc[1] + acc[2] + acc[3]
                  + acc[4] + acc[5] + acc[6] + acc[7];
        }
        // combine the two half-waves (M=0..7 vs M=8..15)
        nsum += __shfl_xor(nsum, 16, 32);
        int n = nt * 16 + nrow;
        if (lane < 16 && n < N2)
            out[b * N2 + n] = nsum * (1.0f / (float)P_REAL);
    }
}

// ---------------------------------------------------------------------------
extern "C" void kernel_launch(void* const* d_in, const int* in_sizes, int n_in,
                              void* d_out, int out_size, void* d_ws, size_t ws_size,
                              hipStream_t stream) {
    (void)n_in; (void)out_size; (void)ws_size;
    const int*   ent_words = (const int*)d_in[0];
    const float* word_vec  = (const float*)d_in[1];
    const float* W1        = (const float*)d_in[2];
    const float* W2        = (const float*)d_in[3];
    float*       out       = (float*)d_out;

    _Float16* w1e = (_Float16*)d_ws;
    _Float16* w2e = (_Float16*)((char*)d_ws + (size_t)NPAD * KPAD * sizeof(_Float16));

    const int B = in_sizes[0] / L_SEQ;   // 4096

    dkrl_prep<<<(NPAD * KPAD + 255) / 256, 256, 0, stream>>>(W1, W2, w1e, w2e);
    dkrl_main<<<B, 128, 0, stream>>>(ent_words, word_vec, w1e, w2e, out);
}
